// Block_6433861009667
// MI455X (gfx1250) — matrix operand order repaired and checked by reading, not compiled, not run on hardware
//
#include <hip/hip_runtime.h>

// ---------------------------------------------------------------------------
// Types / helpers
// ---------------------------------------------------------------------------
typedef __attribute__((ext_vector_type(16))) __bf16 v16bf;
typedef __attribute__((ext_vector_type(8)))  float  v8f;
typedef __attribute__((ext_vector_type(4)))  int    i32x4;

__device__ __forceinline__ unsigned short f2bf(float x) {
  unsigned int u = __float_as_uint(x);
  u += 0x7FFFu + ((u >> 16) & 1u);          // round-to-nearest-even
  return (unsigned short)(u >> 16);
}

union FragU { struct { uint4 lo, hi; } q; v16bf v; };

// A-matrix fragment (16x32 bf16): lane row = lane%16, K chunks per ISA layout:
// half 0 -> K {0..7, 16..23}, half 1 -> K {8..15, 24..31}
__device__ __forceinline__ v16bf ldfragA(const unsigned short* p0, int half) {
  const unsigned short* p = p0 + half * 8;
  FragU f;
  f.q.lo = *(const uint4*)(p);
  f.q.hi = *(const uint4*)(p + 16);
  return f.v;
}
// B-matrix fragment (32x16 bf16): lane col = lane%16, K chunks as A:
// half 0 -> K {0..7, 16..23}, half 1 -> K {8..15, 24..31}
__device__ __forceinline__ v16bf ldfragB(const unsigned short* p0, int half) {
  const unsigned short* p = p0 + half * 8;
  FragU f;
  f.q.lo = *(const uint4*)(p);
  f.q.hi = *(const uint4*)(p + 16);
  return f.v;
}

__device__ __forceinline__ v8f wmma_bf16(v16bf a, v16bf b, v8f c) {
  return __builtin_amdgcn_wmma_f32_16x16x32_bf16(false, a, false, b,
                                                 (short)0, c, false, false);
}

// ---------------------------------------------------------------------------
// Async global->LDS copy path (CDNA5 ASYNCcnt), with synchronous fallback
// ---------------------------------------------------------------------------
#if defined(__gfx1250__) &&                                            \
    __has_builtin(__builtin_amdgcn_global_load_async_to_lds_b128) &&   \
    __has_builtin(__builtin_amdgcn_s_wait_asynccnt)
#define USE_ASYNC_LDS 1
#else
#define USE_ASYNC_LDS 0
#endif

__device__ __forceinline__ void copy16_g2l(unsigned short* lds_dst,
                                           const unsigned short* gsrc) {
#if USE_ASYNC_LDS
  __builtin_amdgcn_global_load_async_to_lds_b128(
      (i32x4*)gsrc, (i32x4*)lds_dst, 0, 0);
#else
  *(uint4*)lds_dst = *(const uint4*)gsrc;
#endif
}

__device__ __forceinline__ void wait_g2l() {
#if USE_ASYNC_LDS
  __builtin_amdgcn_s_wait_asynccnt(0);
#endif
}

// ---------------------------------------------------------------------------
// fp32 -> bf16 convert
// ---------------------------------------------------------------------------
__global__ __launch_bounds__(256) void cvt_kernel(const float* __restrict__ src,
                                                  unsigned short* __restrict__ dst,
                                                  int n) {
  int base = blockIdx.x * 1024 + threadIdx.x;
  #pragma unroll
  for (int j = 0; j < 4; ++j) {
    int k = base + j * 256;
    if (k < n) dst[k] = f2bf(src[k]);
  }
}

// ---------------------------------------------------------------------------
// LayerNorm: fp32 in -> bf16 out (row = one block, C = 1024)
// ---------------------------------------------------------------------------
__global__ __launch_bounds__(256) void ln_kernel(const float* __restrict__ x,
                                                 const float* __restrict__ w,
                                                 const float* __restrict__ b,
                                                 unsigned short* __restrict__ y) {
  __shared__ float red[256];
  const int tid = threadIdx.x;
  const size_t row = blockIdx.x;
  const float* xr = x + row * 1024;

  float vals[4];
  float s = 0.f;
  #pragma unroll
  for (int i = 0; i < 4; ++i) { vals[i] = xr[tid + i * 256]; s += vals[i]; }
  red[tid] = s; __syncthreads();
  for (int o = 128; o > 0; o >>= 1) {
    if (tid < o) red[tid] += red[tid + o];
    __syncthreads();
  }
  const float mu = red[0] * (1.f / 1024.f);
  __syncthreads();

  float vs = 0.f;
  #pragma unroll
  for (int i = 0; i < 4; ++i) { float d = vals[i] - mu; vs += d * d; }
  red[tid] = vs; __syncthreads();
  for (int o = 128; o > 0; o >>= 1) {
    if (tid < o) red[tid] += red[tid + o];
    __syncthreads();
  }
  const float rstd = rsqrtf(red[0] * (1.f / 1024.f) + 1e-5f);

  #pragma unroll
  for (int i = 0; i < 4; ++i) {
    int c = tid + i * 256;
    y[row * 1024 + c] = f2bf((vals[i] - mu) * rstd * w[c] + b[c]);
  }
}

// ---------------------------------------------------------------------------
// Generic WMMA GEMM:  out[M,Nn] = A[M,K](bf16) * Bw[Nn,K]^T(bf16) (+bias)(+act)(+resid)
// Block tile 128x128. K-step 64, double-buffered LDS, async global->LDS
// prefetch of stage i+1 overlapping the 16 WMMAs of stage i; one barrier/stage.
// ---------------------------------------------------------------------------
template<bool GELU, bool OUT_BF16, bool EXPERT>
__global__ __launch_bounds__(256)
void gemm_bf16(const unsigned short* __restrict__ A,
               const unsigned short* __restrict__ Bw,
               const float* __restrict__ bias,
               const float* __restrict__ resid,
               float* __restrict__ outF,
               unsigned short* __restrict__ outH,
               int K, int ldc,
               const long long* __restrict__ indices) {
  const int tid  = threadIdx.x;
  const int lane = tid & 31;
  const int wave = tid >> 5;
  const int half = lane >> 4;
  const int l16  = lane & 15;
  const int wm = wave & 3;   // 4 row groups * 32 rows
  const int wn = wave >> 2;  // 2 col groups * 64 cols

  const int rowBase = blockIdx.y * 128;
  const int colBase = blockIdx.x * 128;
  size_t batchRowOff = 0;

  if constexpr (EXPERT) {
    long long e = indices[blockIdx.z];
    Bw   += (size_t)e * 256 * (size_t)K;     // exp_w[e] : [P, HID]
    bias += (size_t)e * 256;                 // exp_b[e] : [P]
    A    += (size_t)blockIdx.z * 1024 * (size_t)K;
    batchRowOff = (size_t)blockIdx.z * 1024;
  }

  __shared__ unsigned short As[2][128 * 72];
  __shared__ unsigned short Bs[2][128 * 72];

  v8f zero = {};
  v8f acc[2][4];
  #pragma unroll
  for (int i = 0; i < 2; ++i)
    #pragma unroll
    for (int j = 0; j < 4; ++j) acc[i][j] = zero;

  // stage a 128x64 bf16 tile of A and B into buffer bi (1024 16B chunks each)
  auto stage = [&](int kk, int bi) {
    #pragma unroll
    for (int it = 0; it < 4; ++it) {
      int id = tid + it * 256;
      int r = id >> 3;
      int c = (id & 7) * 8;
      copy16_g2l(&As[bi][r * 72 + c], A + (size_t)(rowBase + r) * K + kk + c);
      copy16_g2l(&Bs[bi][r * 72 + c], Bw + (size_t)(colBase + r) * K + kk + c);
    }
  };

  const int steps = K >> 6;  // K / 64
  stage(0, 0);

  for (int i = 0; i < steps; ++i) {
    wait_g2l();            // own stage-i async copies have landed in LDS
    __syncthreads();       // everyone's have; everyone done reading buf[(i+1)&1]
    if (i + 1 < steps) stage((i + 1) << 6, (i + 1) & 1);  // overlaps WMMAs below

    const unsigned short* as = As[i & 1];
    const unsigned short* bs = Bs[i & 1];
    #pragma unroll
    for (int ks = 0; ks < 2; ++ks) {        // two K=32 sub-steps
      v16bf af[2], bf[4];
      #pragma unroll
      for (int mt = 0; mt < 2; ++mt)
        af[mt] = ldfragA(&as[(wm * 32 + mt * 16 + l16) * 72 + ks * 32], half);
      #pragma unroll
      for (int nt = 0; nt < 4; ++nt)
        bf[nt] = ldfragB(&bs[(wn * 64 + nt * 16 + l16) * 72 + ks * 32], half);
      #pragma unroll
      for (int mt = 0; mt < 2; ++mt)
        #pragma unroll
        for (int nt = 0; nt < 4; ++nt)
          acc[mt][nt] = wmma_bf16(af[mt], bf[nt], acc[mt][nt]);
    }
  }

  // epilogue
  #pragma unroll
  for (int mt = 0; mt < 2; ++mt)
    #pragma unroll
    for (int nt = 0; nt < 4; ++nt)
      #pragma unroll
      for (int v = 0; v < 8; ++v) {
        int row = rowBase + wm * 32 + mt * 16 + v + 8 * half;
        int col = colBase + wn * 64 + nt * 16 + l16;
        float o = acc[mt][nt][v];
        if (bias) o += bias[col];
        if constexpr (GELU) o = 0.5f * o * (1.f + erff(o * 0.70710678f));
        size_t oidx = (batchRowOff + (size_t)row) * (size_t)ldc + col;
        if (resid) o += resid[oidx];
        if constexpr (OUT_BF16) outH[oidx] = f2bf(o);
        else                    outF[oidx] = o;
      }
}

// ---------------------------------------------------------------------------
// Flash attention: grid (N/128, H, B); block 256 (8 waves x 16 query rows).
// qkv: bf16 [B, N, 3, H, 64]; out: bf16 [B, N, C]
// ---------------------------------------------------------------------------
__global__ __launch_bounds__(256)
void attn_kernel(const unsigned short* __restrict__ qkv,
                 unsigned short* __restrict__ attn_out) {
  const int lane = threadIdx.x & 31;
  const int wave = threadIdx.x >> 5;
  const int half = lane >> 4;
  const int l16  = lane & 15;
  const int b = blockIdx.z, h = blockIdx.y;
  const int q0 = blockIdx.x * 128 + wave * 16;

  __shared__ unsigned short Kl[64 * 72];          // K chunk, row-major [key][hd]
  __shared__ unsigned short Vt[64 * 72];          // V chunk transposed [hd][key]
  __shared__ unsigned short Pl[8][16 * 64];       // per-wave P tile (bf16)

  // load Q fragments once (HD=64 -> 2 K-chunks of 32), rows q0..q0+15
  v16bf qf[2];
  {
    int n = q0 + l16;
    const unsigned short* qp =
        qkv + ((((size_t)b * 1024 + n) * 3 + 0) * 16 + h) * 64;
    #pragma unroll
    for (int f = 0; f < 2; ++f) qf[f] = ldfragA(qp + f * 32, half);
  }

  v8f zero = {};
  v8f acc[4] = {zero, zero, zero, zero};          // 16x64 output accum
  float m_i[8], l_i[8];
  #pragma unroll
  for (int v = 0; v < 8; ++v) { m_i[v] = -1e30f; l_i[v] = 0.f; }

  for (int kc = 0; kc < 1024; kc += 64) {
    // stage K (async, row-major) and V (transposed via VGPRs): 64x64 bf16 each
    #pragma unroll
    for (int it = 0; it < 2; ++it) {
      int id = threadIdx.x + it * 256;            // 512 chunks of 8 elems
      int row = id >> 3;
      int c   = (id & 7) * 8;
      const unsigned short* kp =
          qkv + ((((size_t)b * 1024 + kc + row) * 3 + 1) * 16 + h) * 64 + c;
      copy16_g2l(&Kl[row * 72 + c], kp);
      const unsigned short* vp =
          qkv + ((((size_t)b * 1024 + kc + row) * 3 + 2) * 16 + h) * 64 + c;
      uint4 vv = *(const uint4*)vp;
      const unsigned short* ve = (const unsigned short*)&vv;
      #pragma unroll
      for (int j = 0; j < 8; ++j) Vt[(c + j) * 72 + row] = ve[j];
    }
    wait_g2l();
    __syncthreads();

    // S = (q * scale) @ k^T : 4 key-tiles of 16, K = HD = 64
    v8f s[4];
    #pragma unroll
    for (int nt = 0; nt < 4; ++nt) {
      v8f c = zero;
      #pragma unroll
      for (int f = 0; f < 2; ++f) {
        v16bf bfr = ldfragB(&Kl[(nt * 16 + l16) * 72 + f * 32], half);
        c = wmma_bf16(qf[f], bfr, c);
      }
      s[nt] = c;
    }
    #pragma unroll
    for (int nt = 0; nt < 4; ++nt)
      #pragma unroll
      for (int v = 0; v < 8; ++v) s[nt][v] *= 0.125f;   // HD^-0.5

    // online softmax (rows never cross waves; 16-lane butterfly per half)
    #pragma unroll
    for (int v = 0; v < 8; ++v) {
      float t = fmaxf(fmaxf(s[0][v], s[1][v]), fmaxf(s[2][v], s[3][v]));
      #pragma unroll
      for (int off = 1; off < 16; off <<= 1) t = fmaxf(t, __shfl_xor(t, off, 32));
      float mnew = fmaxf(m_i[v], t);
      float scale = __expf(m_i[v] - mnew);
      m_i[v] = mnew;
      l_i[v] *= scale;
      #pragma unroll
      for (int j = 0; j < 4; ++j) acc[j][v] *= scale;
      #pragma unroll
      for (int nt = 0; nt < 4; ++nt) s[nt][v] = __expf(s[nt][v] - mnew);
    }
    #pragma unroll
    for (int v = 0; v < 8; ++v) {
      float t = s[0][v] + s[1][v] + s[2][v] + s[3][v];
      #pragma unroll
      for (int off = 1; off < 16; off <<= 1) t += __shfl_xor(t, off, 32);
      l_i[v] += t;
    }

    // repack P (C-layout fp32) -> bf16 A-layout via per-wave LDS tile
    #pragma unroll
    for (int nt = 0; nt < 4; ++nt)
      #pragma unroll
      for (int v = 0; v < 8; ++v)
        Pl[wave][(v + 8 * half) * 64 + nt * 16 + l16] = f2bf(s[nt][v]);

    // O += P @ V : contraction over 64 keys (2 K-chunks), 4 hd-tiles
    #pragma unroll
    for (int f = 0; f < 2; ++f) {
      v16bf af = ldfragA(&Pl[wave][l16 * 64 + f * 32], half);
      #pragma unroll
      for (int dt = 0; dt < 4; ++dt) {
        v16bf bfr = ldfragB(&Vt[(dt * 16 + l16) * 72 + f * 32], half);
        acc[dt] = wmma_bf16(af, bfr, acc[dt]);
      }
    }
    __syncthreads();
  }

  // normalize and write bf16 [B,N,C] at channel h*64
  #pragma unroll
  for (int dt = 0; dt < 4; ++dt)
    #pragma unroll
    for (int v = 0; v < 8; ++v) {
      float o = acc[dt][v] / l_i[v];
      int row = q0 + v + 8 * half;
      int col = h * 64 + dt * 16 + l16;
      attn_out[((size_t)b * 1024 + row) * 1024 + col] = f2bf(o);
    }
}

// ---------------------------------------------------------------------------
// Host-side orchestration
// ---------------------------------------------------------------------------
extern "C" void kernel_launch(void* const* d_in, const int* in_sizes, int n_in,
                              void* d_out, int out_size, void* d_ws, size_t ws_size,
                              hipStream_t stream) {
  (void)in_sizes; (void)n_in; (void)out_size; (void)ws_size;
  constexpr int Bb = 4, Nseq = 1024, Cc = 1024, HID = 4096, SH = 768;
  constexpr size_t M = (size_t)Bb * Nseq;   // 4096 rows

  const float*      x      = (const float*)d_in[0];
  const long long*  idx    = (const long long*)d_in[1];
  const float*      ln1_w  = (const float*)d_in[2];
  const float*      ln1_b  = (const float*)d_in[3];
  const float*      qkv_w  = (const float*)d_in[4];
  const float*      proj_w = (const float*)d_in[5];
  const float*      proj_b = (const float*)d_in[6];
  const float*      ln2_w  = (const float*)d_in[7];
  const float*      ln2_b  = (const float*)d_in[8];
  const float*      fc1_w  = (const float*)d_in[9];
  const float*      fc1_b  = (const float*)d_in[10];
  const float*      fc2_w  = (const float*)d_in[11];
  const float*      fc2_b  = (const float*)d_in[12];
  const float*      exp_w  = (const float*)d_in[13];
  const float*      exp_b  = (const float*)d_in[14];
  float*            out    = (float*)d_out;

  // workspace carve-up (bytes, 256-aligned)
  char* base = (char*)d_ws;
  size_t off = 0;
  auto carve = [&](size_t bytes) -> char* {
    char* p = base + off;
    off = (off + bytes + 255) & ~(size_t)255;
    return p;
  };
  unsigned short* wqkv  = (unsigned short*)carve((size_t)3 * Cc * Cc * 2);
  unsigned short* wproj = (unsigned short*)carve((size_t)Cc * Cc * 2);
  unsigned short* wfc1  = (unsigned short*)carve((size_t)HID * Cc * 2);
  unsigned short* wfc2  = (unsigned short*)carve((size_t)SH * HID * 2);
  unsigned short* wexp  = (unsigned short*)carve((size_t)8 * 256 * HID * 2);
  unsigned short* xln1  = (unsigned short*)carve(M * Cc * 2);
  unsigned short* qkvb  = (unsigned short*)carve(M * 3 * Cc * 2);
  unsigned short* attnb = (unsigned short*)carve(M * Cc * 2);
  float*          x1    = (float*)carve(M * Cc * 4);
  unsigned short* x2ln  = (unsigned short*)carve(M * Cc * 2);
  unsigned short* hbuf  = (unsigned short*)carve(M * HID * 2);

  auto cvt = [&](const float* s, unsigned short* d, int n) {
    cvt_kernel<<<(n + 1023) / 1024, 256, 0, stream>>>(s, d, n);
  };
  cvt(qkv_w,  wqkv,  3 * Cc * Cc);
  cvt(proj_w, wproj, Cc * Cc);
  cvt(fc1_w,  wfc1,  HID * Cc);
  cvt(fc2_w,  wfc2,  SH * HID);
  cvt(exp_w,  wexp,  8 * 256 * HID);

  // ln1
  ln_kernel<<<(int)M, 256, 0, stream>>>(x, ln1_w, ln1_b, xln1);

  // qkv = ln1(x) @ qkv_w^T  -> bf16 [M, 3C]
  gemm_bf16<false, true, false><<<dim3(3 * Cc / 128, M / 128, 1), 256, 0, stream>>>(
      xln1, wqkv, nullptr, nullptr, nullptr, qkvb, Cc, 3 * Cc, nullptr);

  // attention -> bf16 [M, C]
  attn_kernel<<<dim3(Nseq / 128, 16, Bb), 256, 0, stream>>>(qkvb, attnb);

  // x1 = x + attn @ proj_w^T + proj_b   (fp32)
  gemm_bf16<false, false, false><<<dim3(Cc / 128, M / 128, 1), 256, 0, stream>>>(
      attnb, wproj, proj_b, x, x1, nullptr, Cc, Cc, nullptr);

  // ln2
  ln_kernel<<<(int)M, 256, 0, stream>>>(x1, ln2_w, ln2_b, x2ln);

  // h = gelu(ln2 @ fc1_w^T + fc1_b) -> bf16 [M, HID]
  gemm_bf16<true, true, false><<<dim3(HID / 128, M / 128, 1), 256, 0, stream>>>(
      x2ln, wfc1, fc1_b, nullptr, nullptr, hbuf, Cc, HID, nullptr);

  // out[:, :768] = x1[:, :768] + h @ fc2_w^T + fc2_b
  gemm_bf16<false, false, false><<<dim3(SH / 128, M / 128, 1), 256, 0, stream>>>(
      hbuf, wfc2, fc2_b, x1, out, nullptr, HID, Cc, nullptr);

  // out[:, 768:] = x1[:, 768:] + h @ exp_w[idx[b]]^T + exp_b[idx[b]]
  gemm_bf16<false, false, true><<<dim3(256 / 128, Nseq / 128, Bb), 256, 0, stream>>>(
      hbuf, wexp, exp_b, x1 + SH, out + SH, nullptr, HID, Cc, idx);
}
